// MultiScaleDeformAttn_84997402788104
// MI455X (gfx1250) — matrix operand order repaired and checked by reading, not compiled, not run on hardware
//
#include <hip/hip_runtime.h>
#include <math.h>

// ---------------------------------------------------------------------------
// Multi-Scale Deformable Attention (encoder) for MI455X / gfx1250.
// GEMMs run on v_wmma_f32_16x16x32_bf16 (bf16 in, f32 accumulate),
// register-blocked 1x4 over N tiles so each A fragment feeds 4 WMMAs.
#define DM     256          // d_model
#define NH     8            // heads
#define NL     4            // levels
#define NP     4            // points
#define HD     32           // head dim == wave32 lanes
#define NQ_REF 13294

typedef __attribute__((ext_vector_type(16))) __bf16 v16bf;
typedef __attribute__((ext_vector_type(8)))  __bf16 v8bf;
typedef __attribute__((ext_vector_type(8)))  float  v8f;

// ---------------------------------------------------------------------------
// Operand loaders: build the 16x32 A fragment in contiguous k order, matching B
// (lane<16: K=0..15 of the k-step; lane>=16: K=16..31).
__device__ __forceinline__ v16bf loadA(const float* p0, const float* p1) {
    const float4 f0 = *reinterpret_cast<const float4*>(p0);
    const float4 f1 = *reinterpret_cast<const float4*>(p0 + 4);
    const float4 f2 = *reinterpret_cast<const float4*>(p1);
    const float4 f3 = *reinterpret_cast<const float4*>(p1 + 4);
    v16bf a;
    a[0]=(__bf16)f0.x;  a[1]=(__bf16)f0.y;  a[2]=(__bf16)f0.z;  a[3]=(__bf16)f0.w;
    a[4]=(__bf16)f1.x;  a[5]=(__bf16)f1.y;  a[6]=(__bf16)f1.z;  a[7]=(__bf16)f1.w;
    a[8]=(__bf16)f2.x;  a[9]=(__bf16)f2.y;  a[10]=(__bf16)f2.z; a[11]=(__bf16)f2.w;
    a[12]=(__bf16)f3.x; a[13]=(__bf16)f3.y; a[14]=(__bf16)f3.z; a[15]=(__bf16)f3.w;
    return a;
}
__device__ __forceinline__ v16bf loadA(const __bf16* p0, const __bf16* p1) {
    const v8bf lo = *reinterpret_cast<const v8bf*>(p0);
    const v8bf hi = *reinterpret_cast<const v8bf*>(p1);
    v16bf a;
    a[0]=lo[0]; a[1]=lo[1]; a[2]=lo[2];  a[3]=lo[3];
    a[4]=lo[4]; a[5]=lo[5]; a[6]=lo[6];  a[7]=lo[7];
    a[8]=hi[0]; a[9]=hi[1]; a[10]=hi[2]; a[11]=hi[3];
    a[12]=hi[4];a[13]=hi[5];a[14]=hi[6]; a[15]=hi[7];
    return a;
}
__device__ __forceinline__ void storeOut(float*  C, size_t i, float v) { C[i] = v; }
__device__ __forceinline__ void storeOut(__bf16* C, size_t i, float v) { C[i] = (__bf16)v; }

// ---------------------------------------------------------------------------
// Tiled WMMA GEMM: C[M,N] = A[M,K] * Bt[N,K]^T + bias[N].
// Block = 4 waves; each wave owns one 16-row M tile and NTW 16-col N tiles,
// reusing its A fragment across NTW WMMAs per k-step (4x arithmetic
// intensity vs one-tile-per-wave).
template <typename AT, typename OT, int NTW>
__global__ void __launch_bounds__(128)
gemm_wmma_bf16(const AT* __restrict__ A, const __bf16* __restrict__ Bt,
               const float* __restrict__ bias, OT* __restrict__ C,
               int M, int N, int K) {
    const int wave = threadIdx.x >> 5;      // 0..3
    const int lane = threadIdx.x & 31;
    const int mt   = blockIdx.x;
    const int nt0  = wave * NTW;            // first N tile for this wave
    if (nt0 * 16 >= N) return;

    const int hl = lane >> 4;               // wave half
    const int mr = lane & 15;
    int row = mt * 16 + mr;
    if (row >= M) row = M - 1;              // clamp loads on the ragged tile
    const int coll = lane & 15;
    const AT* Arow = A + (size_t)row * K;

    v8f acc[NTW] = {};
    for (int k0 = 0; k0 < K; k0 += 32) {
        if (k0 + 32 < K) __builtin_prefetch(Arow + k0 + 32, 0, 3);
        // A fragment: 16x32 bf16, contiguous k per lane half (loaded ONCE per k-step)
        const v16bf a = loadA(Arow + k0 + hl * 16, Arow + k0 + hl * 16 + 8);
        #pragma unroll
        for (int j = 0; j < NTW; ++j) {
            // B fragment: 32x16 bf16; lane<16 holds K=k0..k0+15 of its column,
            // lane>=16 holds K=k0+16..k0+31 (contiguous since Bt is [N,K]).
            const __bf16* Brow = Bt + (size_t)((nt0 + j) * 16 + coll) * K;
            const v16bf b = *reinterpret_cast<const v16bf*>(Brow + k0 + hl * 16);
            acc[j] = __builtin_amdgcn_wmma_f32_16x16x32_bf16(
                         false, a, false, b, (short)0, acc[j], false, false);
        }
    }

    #pragma unroll
    for (int j = 0; j < NTW; ++j) {
        const int   bcol = (nt0 + j) * 16 + coll;
        const float bb   = bias[bcol];
        #pragma unroll
        for (int r = 0; r < 8; ++r) {       // VGPR r holds M = r (+8, upper half)
            const int orow = mt * 16 + hl * 8 + r;
            if (orow < M) storeOut(C, (size_t)orow * N + bcol, acc[j][r] + bb);
        }
    }
}

// ---------------------------------------------------------------------------
// Weight prep: transpose + convert fp32 -> bf16 so WMMA B loads are contiguous.
__global__ void prep_weights(const float* __restrict__ Wso, const float* __restrict__ Waw,
                             const float* __restrict__ Wv,  const float* __restrict__ Wo,
                             __bf16* __restrict__ WsoT, __bf16* __restrict__ WawT,
                             __bf16* __restrict__ WvT,  __bf16* __restrict__ WoT) {
    const int idx = blockIdx.x * blockDim.x + threadIdx.x;  // 0..65535
    const int k = idx & (DM - 1);
    const int n = idx >> 8;
    WsoT[(size_t)n * DM + k] = (__bf16)Wso[(size_t)k * DM + n];
    WvT [(size_t)n * DM + k] = (__bf16)Wv [(size_t)k * DM + n];
    WoT [(size_t)n * DM + k] = (__bf16)Wo [(size_t)k * DM + n];
    if (n < NH * NL * NP)
        WawT[(size_t)n * DM + k] = (__bf16)Waw[(size_t)k * (NH * NL * NP) + n];
}

// ---------------------------------------------------------------------------
// Sampling stage: one wave per (b*Nq+q, head); lane == channel d (HD==32).
// Fused softmax over the 16 logits + bilinear gather (zero padding) from the
// bf16 value_proj tensor; writes bf16 activations for the output-proj GEMM.
__global__ void __launch_bounds__(256)
msda_sample(const float* __restrict__ ref, const __bf16* __restrict__ vproj,
            const float* __restrict__ off, const float* __restrict__ awl,
            __bf16* __restrict__ attn, int Nq) {
    constexpr int lw_c[NL]     = {100, 50, 25, 13};
    constexpr int lh_c[NL]     = {100, 50, 25, 13};
    constexpr int starts_c[NL] = {0, 10000, 12500, 13125};

    const int row = blockIdx.x;           // b*Nq + q
    const int h   = threadIdx.x >> 5;
    const int d   = threadIdx.x & 31;
    const int b   = row / Nq;
    const int vb  = b * Nq;               // Nv == Nq (encoder)

    // softmax over this head's 16 (level,point) logits — broadcast loads
    const float* wl = awl + (size_t)row * (NH * NL * NP) + h * (NL * NP);
    float lg[16];
    #pragma unroll
    for (int i = 0; i < 4; ++i) {
        const float4 t = reinterpret_cast<const float4*>(wl)[i];
        lg[4*i+0]=t.x; lg[4*i+1]=t.y; lg[4*i+2]=t.z; lg[4*i+3]=t.w;
    }
    float mx = lg[0];
    #pragma unroll
    for (int i = 1; i < 16; ++i) mx = fmaxf(mx, lg[i]);
    float s = 0.f;
    #pragma unroll
    for (int i = 0; i < 16; ++i) { lg[i] = __expf(lg[i] - mx); s += lg[i]; }
    const float inv = 1.f / s;

    const float* offr = off + (size_t)row * DM + h * (NL * NP * 2);
    float acc = 0.f;
    #pragma unroll
    for (int l = 0; l < NL; ++l) {
        const int   lww = lw_c[l], lhh = lh_c[l];
        const float rx  = ref[((size_t)row * NL + l) * 2 + 0];
        const float ry  = ref[((size_t)row * NL + l) * 2 + 1];
        const int   base = vb + starts_c[l];
        #pragma unroll
        for (int p = 0; p < NP; ++p) {
            const float ox = offr[(l * NP + p) * 2 + 0];
            const float oy = offr[(l * NP + p) * 2 + 1];
            // gx = ((2*loc_x - 1) + 1)*W/2 - 0.5 = ref_x*W + off_x - 0.5
            const float gx = rx * (float)lww + ox - 0.5f;
            const float gy = ry * (float)lhh + oy - 0.5f;
            const float fx = floorf(gx), fy = floorf(gy);
            const float wx = gx - fx,    wy = gy - fy;
            const int   x0 = (int)fx,    y0 = (int)fy;
            const float wgt = lg[l * NP + p] * inv;

            float v00 = 0.f, v10 = 0.f, v01 = 0.f, v11 = 0.f;
            const bool xi0 = (x0 >= 0)     & (x0 < lww);
            const bool xi1 = (x0 + 1 >= 0) & (x0 + 1 < lww);
            const bool yi0 = (y0 >= 0)     & (y0 < lhh);
            const bool yi1 = (y0 + 1 >= 0) & (y0 + 1 < lhh);
            const size_t ch = (size_t)h * HD + d;
            if (xi0 & yi0) v00 = (float)vproj[((size_t)(base + y0*lww + x0))     * DM + ch];
            if (xi1 & yi0) v10 = (float)vproj[((size_t)(base + y0*lww + x0 + 1)) * DM + ch];
            if (xi0 & yi1) v01 = (float)vproj[((size_t)(base + (y0+1)*lww + x0)) * DM + ch];
            if (xi1 & yi1) v11 = (float)vproj[((size_t)(base + (y0+1)*lww + x0+1))*DM + ch];
            acc += wgt * ((v00*(1.f-wx) + v10*wx)*(1.f-wy) + (v01*(1.f-wx) + v11*wx)*wy);
        }
    }
    attn[(size_t)row * DM + h * HD + d] = (__bf16)acc;
}

// ---------------------------------------------------------------------------
extern "C" void kernel_launch(void* const* d_in, const int* in_sizes, int n_in,
                              void* d_out, int out_size, void* d_ws, size_t ws_size,
                              hipStream_t stream) {
    (void)n_in; (void)out_size; (void)ws_size;
    const float* query = (const float*)d_in[0];
    const float* ref   = (const float*)d_in[1];
    const float* value = (const float*)d_in[2];
    const float* Wso   = (const float*)d_in[4];
    const float* bso   = (const float*)d_in[5];
    const float* Waw   = (const float*)d_in[6];
    const float* baw   = (const float*)d_in[7];
    const float* Wv    = (const float*)d_in[8];
    const float* bv    = (const float*)d_in[9];
    const float* Wo    = (const float*)d_in[10];
    const float* bo    = (const float*)d_in[11];

    const int Nq  = NQ_REF;
    const int M   = in_sizes[0] / DM;    // B*Nq rows
    const int NAW = NH * NL * NP;        // 128

    char*  ws = (char*)d_ws;
    size_t o  = 0;
    auto take = [&](size_t bytes) -> char* {
        o = (o + 255) & ~(size_t)255;
        char* p = ws + o; o += bytes; return p;
    };
    __bf16* WsoT  = (__bf16*)take((size_t)DM * DM * 2);
    __bf16* WawT  = (__bf16*)take((size_t)NAW * DM * 2);
    __bf16* WvT   = (__bf16*)take((size_t)DM * DM * 2);
    __bf16* WoT   = (__bf16*)take((size_t)DM * DM * 2);
    __bf16* vproj = (__bf16*)take((size_t)M * DM * 2);
    float*  offb  = (float*) take((size_t)M * DM * 4);
    float*  awb   = (float*) take((size_t)M * NAW * 4);
    __bf16* attnb = (__bf16*)take((size_t)M * DM * 2);

    prep_weights<<<DM, DM, 0, stream>>>(Wso, Waw, Wv, Wo, WsoT, WawT, WvT, WoT);

    const int Mt = (M + 15) / 16;
    // value_proj -> bf16 activations for the gather stage (N=256: 4 tiles/wave)
    gemm_wmma_bf16<float, __bf16, 4><<<Mt, 128, 0, stream>>>(
        value, WvT, bv, vproj, M, DM, DM);
    // sampling offsets (fp32 out — pixel coords need full precision)
    gemm_wmma_bf16<float, float, 4><<<Mt, 128, 0, stream>>>(
        query, WsoT, bso, offb, M, DM, DM);
    // attention-weight logits (N=128: 2 tiles/wave)
    gemm_wmma_bf16<float, float, 2><<<Mt, 128, 0, stream>>>(
        query, WawT, baw, awb, M, NAW, DM);
    // softmax + bilinear gather + weighted sum
    msda_sample<<<M, 256, 0, stream>>>(ref, vproj, offb, awb, attnb, Nq);
    // output projection straight into d_out
    gemm_wmma_bf16<__bf16, float, 4><<<Mt, 128, 0, stream>>>(
        attnb, WoT, bo, (float*)d_out, M, DM, DM);
}
